// MiniGPT_38182259262112
// MI455X (gfx1250) — matrix operand order repaired and checked by reading, not compiled, not run on hardware
//
#include <hip/hip_runtime.h>
#include <hip/hip_bf16.h>

// ---------------------------------------------------------------------------
// MiniGPT forward for MI455X (gfx1250, wave32, WMMA + TDM).
// All GEMMs run on v_wmma_f32_16x16x32_f16 (f16 in, fp32 accumulate).
// f16 activation tiles are staged to LDS via the Tensor Data Mover
// (tensor_load_to_lds + s_wait_tensorcnt) when the builtin is available;
// fp32 weights are converted to f16 on the fly while staging to LDS.
// ---------------------------------------------------------------------------

typedef _Float16 v16h __attribute__((ext_vector_type(16)));
typedef _Float16 v8h  __attribute__((ext_vector_type(8)));
typedef float    v8f  __attribute__((ext_vector_type(8)));

union FragU { v16h v; v8h h[2]; };

#if __has_builtin(__builtin_amdgcn_tensor_load_to_lds) && \
    __has_builtin(__builtin_amdgcn_s_wait_tensorcnt)
#define USE_TDM 1
#endif

#ifdef USE_TDM
typedef unsigned int u32x4 __attribute__((ext_vector_type(4)));
typedef int          i32x4 __attribute__((ext_vector_type(4)));
typedef int          i32x8 __attribute__((ext_vector_type(8)));

// Issue a 2D f16 tile DMA: [tile_rows x tile_cols] elements from a row-major
// tensor (row stride = ld elements) into LDS at byte offset lds_off.
// LDS destination rows are padded: after every (1<<(interval_code+1)) DWORDs
// written, (pad_code+1) DWORDs are skipped (D# pad feature).
__device__ __forceinline__ void tdm_load_tile_f16(
    const _Float16* gsrc, unsigned lds_off, int tile_cols, int tile_rows,
    int ld, int tensor_rows, int interval_code, int pad_code) {
  unsigned long long ga = (unsigned long long)(size_t)gsrc;
  u32x4 g0;
  g0[0] = 1u;                                    // count=1 (valid user D#)
  g0[1] = lds_off;                               // lds_addr (bytes)
  g0[2] = (unsigned)(ga & 0xffffffffu);          // global_addr[31:0]
  g0[3] = (unsigned)((ga >> 32) & 0x01ffffffu)   // global_addr[56:32]
          | (2u << 30);                          // type = 2 ("image")
  int pad_en = (pad_code >= 0) ? 1 : 0;
  unsigned w0 = (1u << 16)                       // data_size = 1 -> 2 bytes
              | ((unsigned)pad_en << 20)         // pad_enable
              | ((pad_en ? (unsigned)interval_code : 0u) << 22)
              | ((pad_en ? (unsigned)pad_code : 0u) << 25);
  unsigned td0 = (unsigned)ld;                   // tensor_dim0 (elements/row)
  unsigned td1 = (unsigned)tensor_rows;          // tensor_dim1 (rows)
  i32x8 g1;
  g1[0] = (int)w0;
  g1[1] = (int)((td0 & 0xffffu) << 16);                    // dim0 @ bit48
  g1[2] = (int)((td0 >> 16) | ((td1 & 0xffffu) << 16));    // dim0 hi, dim1 lo
  g1[3] = (int)((td1 >> 16) | ((unsigned)tile_cols << 16));// dim1 hi, tile_dim0
  g1[4] = (int)(unsigned)tile_rows;                        // tile_dim1 (dim2=0)
  g1[5] = (int)(unsigned)ld;                               // dim0_stride lo
  g1[6] = 0;                                               // stride hi / dim1_stride
  g1[7] = 0;
  i32x4 z4 = {0, 0, 0, 0};
#if __clang_major__ >= 23
  i32x8 z8 = {0, 0, 0, 0, 0, 0, 0, 0};
  __builtin_amdgcn_tensor_load_to_lds(g0, g1, z4, z4, z8, 0);
#else
  __builtin_amdgcn_tensor_load_to_lds(g0, g1, z4, z4, 0);
#endif
}
#endif  // USE_TDM

__device__ __forceinline__ v8f wmma_f16(v16h a, v16h b, v8f c) {
  // D = A(16x32 f16) * B(32x16 f16) + C(16x16 f32)
  return __builtin_amdgcn_wmma_f32_16x16x32_f16(
      /*neg_a=*/false, a, /*neg_b=*/false, b,
      /*c_mod=*/(short)0, c, /*reuse_a=*/false, /*reuse_b=*/false);
}

// A-matrix fragment (16x32, f16): lanes 0-15 rows, lane half selects K bytes.
// LDS layout: row-major [16+][stride], 16B chunks at k = {h*8, 16+h*8}.
__device__ __forceinline__ v16h frag_a16(const _Float16* p, int stride, int lane) {
  int row = lane & 15, hh = lane >> 4;
  const _Float16* q = p + row * stride + hh * 8;
  FragU f;
  f.h[0] = *(const v8h*)(q);
  f.h[1] = *(const v8h*)(q + 16);
  return f.v;
}

// B-matrix fragment (32x16, f16) from transposed LDS tile [N][K]:
// lane half selects K chunks at k = {h*8, 16+h*8} (same order as frag_a16).
__device__ __forceinline__ v16h frag_b16(const _Float16* p, int stride, int lane) {
  int col = lane & 15, hh = lane >> 4;
  const _Float16* q = p + col * stride + hh * 8;
  FragU f;
  f.h[0] = *(const v8h*)(q);
  f.h[1] = *(const v8h*)(q + 16);
  return f.v;
}

// ---------------------------------------------------------------------------
// Embedding: x[b*T+t, :] = tok_emb[idx[b,t], :] + pos_emb[t, :]
// ---------------------------------------------------------------------------
__global__ __launch_bounds__(256) void embed_kernel(
    const int* __restrict__ idx, const float* __restrict__ tok,
    const float* __restrict__ pos, float* __restrict__ x, int T, int Dd) {
  int row = blockIdx.x;
  int t = row % T;
  int token = idx[row];
  int c = threadIdx.x * 4;
  float4 a = *(const float4*)(tok + (size_t)token * Dd + c);
  float4 p = *(const float4*)(pos + (size_t)t * Dd + c);
  float4 r;
  r.x = a.x + p.x; r.y = a.y + p.y; r.z = a.z + p.z; r.w = a.w + p.w;
  *(float4*)(x + (size_t)row * Dd + c) = r;
}

// ---------------------------------------------------------------------------
// LayerNorm: one row (D=1024) per 256-thread block; f16 output for WMMA A.
// ---------------------------------------------------------------------------
__global__ __launch_bounds__(256) void layernorm_kernel(
    const float* __restrict__ X, const float* __restrict__ g,
    const float* __restrict__ be, _Float16* __restrict__ Y, int Dd) {
  __shared__ float red[16];
  int row = blockIdx.x, tid = threadIdx.x;
  int lane = tid & 31, wave = tid >> 5;
  const float* x = X + (size_t)row * Dd;
  float4 v = *(const float4*)(x + tid * 4);
  float s  = v.x + v.y + v.z + v.w;
  float ss = v.x * v.x + v.y * v.y + v.z * v.z + v.w * v.w;
#pragma unroll
  for (int m = 1; m < 32; m <<= 1) {
    s  += __shfl_xor(s, m, 32);
    ss += __shfl_xor(ss, m, 32);
  }
  if (lane == 0) { red[wave] = s; red[wave + 8] = ss; }
  __syncthreads();
  float stot = 0.f, sstot = 0.f;
#pragma unroll
  for (int i = 0; i < 8; ++i) { stot += red[i]; sstot += red[i + 8]; }
  float mu = stot / Dd;
  float var = sstot / Dd - mu * mu;
  float rstd = rsqrtf(var + 1e-5f);
  _Float16* y = Y + (size_t)row * Dd;
  float xv[4] = {v.x, v.y, v.z, v.w};
#pragma unroll
  for (int j = 0; j < 4; ++j) {
    int c = tid * 4 + j;
    y[c] = (_Float16)((xv[j] - mu) * rstd * g[c] + be[c]);
  }
}

// ---------------------------------------------------------------------------
// Generic WMMA GEMM:  C[M,N] = A_h[M,K](f16) * B[K,N](fp32 weights, cvt->f16)
// Block tile 128x128, 8 waves (2x4) of 64x32, K-step 32.
// A tile staged by TDM (f16 copy with padded LDS rows); B tile staged
// manually (needs fp32->f16 convert + transpose, which TDM cannot do).
// HEADED: B is [H, K, 64] (per-head QKV weight), logical col n = h*64+e.
// Epilogue: +bias, +residual(fp32, ld N), GELU(exact), store f16 or f32.
// ---------------------------------------------------------------------------
template <bool HEADED, bool BIAS, bool RES, bool GELU, bool HALFOUT>
__global__ __launch_bounds__(256) void gemm_wmma(
    const _Float16* __restrict__ A, const float* __restrict__ B,
    const float* __restrict__ bias, const float* __restrict__ res,
    float* __restrict__ Cf, _Float16* __restrict__ Ch,
    int M, int N, int K) {
  __shared__ __align__(16) _Float16 As[128][40];  // [M][K] +pad (80B rows)
  __shared__ __align__(16) _Float16 Bs[128][40];  // [N][K] transposed +pad
  int tid = threadIdx.x;
  int lane = tid & 31, wave = tid >> 5;
  int wm = wave >> 2, wn = wave & 3;
  int bm = blockIdx.x * 128, bn = blockIdx.y * 128;

  v8f acc[4][2] = {};

  for (int k0 = 0; k0 < K; k0 += 32) {
    __syncthreads();
#ifdef USE_TDM
    // Stage A via Tensor Data Mover: 128x32 f16 tile, row = 16 DWORDs,
    // +4 DWORDs LDS pad per row (interval code 3 = 16 DW, pad code 3 = 4 DW).
    if (tid == 0)
      tdm_load_tile_f16(A + (size_t)bm * K + k0,
                        (unsigned)(size_t)&As[0][0],
                        /*tile_cols=*/32, /*tile_rows=*/128,
                        /*ld=*/K, /*tensor_rows=*/M,
                        /*interval_code=*/3, /*pad_code=*/3);
#else
#pragma unroll
    for (int i = 0; i < 2; ++i) {
      int c = tid + i * 256;
      int row = c >> 2, kc = (c & 3) * 8;
      *(v8h*)&As[row][kc] =
          *(const v8h*)(A + (size_t)(bm + row) * K + k0 + kc);
    }
#endif
    // Stage B: 32x128 fp32 -> f16 transposed into LDS.
#pragma unroll
    for (int i = 0; i < 4; ++i) {
      int s = tid + i * 256;
      int kk = s >> 5, n4 = (s & 31) * 4;
      int gn = bn + n4;
      size_t off;
      if (HEADED)
        off = (size_t)(gn >> 6) * (size_t)K * 64 + (size_t)(k0 + kk) * 64 +
              (gn & 63);
      else
        off = (size_t)(k0 + kk) * N + gn;
      float4 w = *(const float4*)(B + off);
      Bs[n4 + 0][kk] = (_Float16)w.x;
      Bs[n4 + 1][kk] = (_Float16)w.y;
      Bs[n4 + 2][kk] = (_Float16)w.z;
      Bs[n4 + 3][kk] = (_Float16)w.w;
      // Prefetch next weight tile into caches (global_prefetch_b8).
      if (i == 0 && k0 + 32 < K) {
        size_t offn;
        if (HEADED)
          offn = (size_t)(gn >> 6) * (size_t)K * 64 +
                 (size_t)(k0 + 32 + kk) * 64 + (gn & 63);
        else
          offn = (size_t)(k0 + 32 + kk) * N + gn;
        __builtin_prefetch(B + offn, 0, 1);
      }
    }
#ifdef USE_TDM
    if (wave == 0) __builtin_amdgcn_s_wait_tensorcnt(0);
#endif
    __syncthreads();

    v16h bf[2];
#pragma unroll
    for (int ns = 0; ns < 2; ++ns)
      bf[ns] = frag_b16(&Bs[wn * 32 + ns * 16][0], 40, lane);
#pragma unroll
    for (int ms = 0; ms < 4; ++ms) {
      v16h af = frag_a16(&As[wm * 64 + ms * 16][0], 40, lane);
      acc[ms][0] = wmma_f16(af, bf[0], acc[ms][0]);
      acc[ms][1] = wmma_f16(af, bf[1], acc[ms][1]);
    }
  }

  // Epilogue. C/D layout: VGPR r -> row (r + 8*(lane>>4)), col = lane&15.
  int col = lane & 15, hh = lane >> 4;
#pragma unroll
  for (int ms = 0; ms < 4; ++ms) {
#pragma unroll
    for (int ns = 0; ns < 2; ++ns) {
      int gc = bn + wn * 32 + ns * 16 + col;
      int rbase = bm + wm * 64 + ms * 16 + hh * 8;
      float bv = BIAS ? bias[gc] : 0.f;
#pragma unroll
      for (int r = 0; r < 8; ++r) {
        int gr = rbase + r;
        float v = acc[ms][ns][r] + bv;
        if (RES) v += res[(size_t)gr * N + gc];
        if (GELU) v = 0.5f * v * (1.f + erff(v * 0.70710678118f));
        if (HALFOUT)
          Ch[(size_t)gr * N + gc] = (_Float16)v;
        else
          Cf[(size_t)gr * N + gc] = v;
      }
    }
  }
}

// ---------------------------------------------------------------------------
// Flash attention (causal), one (b, h, 128-query block) per workgroup.
// S = Q*K^T via WMMA (contract HD=64 = two k-steps), online softmax with
// 16-lane shfl row reductions, O += P*V via WMMA (P staged per-wave in LDS).
// Q and K tiles staged by TDM; V staged manually (needs transpose).
// ---------------------------------------------------------------------------
__global__ __launch_bounds__(256) void attention_kernel(
    const _Float16* __restrict__ Qh, const _Float16* __restrict__ Kh,
    const _Float16* __restrict__ Vh, _Float16* __restrict__ Oh,
    int T, int Dd) {
  __shared__ __align__(16) _Float16 Qs[128][72];      // [q][hd]
  __shared__ __align__(16) _Float16 Ks[64][72];       // [key][hd] (= B^T tile)
  __shared__ __align__(16) _Float16 Vs[64][72];       // [hd][key] (= B^T tile)
  __shared__ __align__(16) _Float16 Ps[8][16][72];    // per-wave P tile [q][key]

  const int HD = 64;
  int tid = threadIdx.x, lane = tid & 31, wave = tid >> 5;
  int col = lane & 15, hh = lane >> 4;
  int qblk = blockIdx.x, h = blockIdx.y, b = blockIdx.z;
  int qbase = qblk * 128;
  size_t rowbase = (size_t)b * T;

#ifdef USE_TDM
  // Q tile: 128 rows x 64 halfs; LDS row = 32 DW + 4 DW pad (codes 4, 3).
  if (tid == 0)
    tdm_load_tile_f16(Qh + (rowbase + qbase) * Dd + h * HD,
                      (unsigned)(size_t)&Qs[0][0],
                      /*tile_cols=*/64, /*tile_rows=*/128,
                      /*ld=*/Dd, /*tensor_rows=*/(int)(4 * T),
                      /*interval_code=*/4, /*pad_code=*/3);
#else
  for (int i = tid; i < 128 * 8; i += 256) {
    int row = i >> 3, cc = (i & 7) * 8;
    *(v8h*)&Qs[row][cc] =
        *(const v8h*)(Qh + (rowbase + qbase + row) * Dd + h * HD + cc);
  }
#endif

  float m[8], l[8];
#pragma unroll
  for (int r = 0; r < 8; ++r) { m[r] = -1e30f; l[r] = 0.f; }
  v8f o[4] = {};

  int nkb = (qbase + 128) / 64;  // causal: only key blocks <= query block
  for (int kb = 0; kb < nkb; ++kb) {
    int kstart = kb * 64;
    __syncthreads();
#ifdef USE_TDM
    if (tid == 0)
      tdm_load_tile_f16(Kh + (rowbase + kstart) * Dd + h * HD,
                        (unsigned)(size_t)&Ks[0][0],
                        /*tile_cols=*/64, /*tile_rows=*/64,
                        /*ld=*/Dd, /*tensor_rows=*/(int)(4 * T),
                        /*interval_code=*/4, /*pad_code=*/3);
#else
    for (int i = tid; i < 64 * 8; i += 256) {
      int row = i >> 3, cc = (i & 7) * 8;
      *(v8h*)&Ks[row][cc] =
          *(const v8h*)(Kh + (rowbase + kstart + row) * Dd + h * HD + cc);
    }
#endif
    for (int i = tid; i < 64 * 8; i += 256) {
      int row = i >> 3, cc = (i & 7) * 8;  // row = key, cc = hd chunk
      v8h vv = *(const v8h*)(Vh + (rowbase + kstart + row) * Dd + h * HD + cc);
#pragma unroll
      for (int j = 0; j < 8; ++j) Vs[cc + j][row] = vv[j];
    }
#ifdef USE_TDM
    if (wave == 0) __builtin_amdgcn_s_wait_tensorcnt(0);  // Q (1st iter) + K
#endif
    __syncthreads();

    // S tiles: 16 q-rows x (4 x 16 keys), contract over HD=64.
    v8f s[4];
#pragma unroll
    for (int ns = 0; ns < 4; ++ns) {
      v8f z = {};
#pragma unroll
      for (int kk = 0; kk < 2; ++kk) {
        v16h a = frag_a16(&Qs[wave * 16][kk * 32], 72, lane);
        v16h bb = frag_b16(&Ks[ns * 16][kk * 32], 72, lane);
        z = wmma_f16(a, bb, z);
      }
      s[ns] = z;
    }

    const float scale = 0.125f;  // HD^-0.5
    float bmax[8];
#pragma unroll
    for (int r = 0; r < 8; ++r) {
      int qrow = qbase + wave * 16 + r + 8 * hh;
      float mx = -1e30f;
#pragma unroll
      for (int ns = 0; ns < 4; ++ns) {
        int kcol = kstart + ns * 16 + col;
        float v = s[ns][r] * scale;
        v = (kcol <= qrow) ? v : -1e30f;
        s[ns][r] = v;
        mx = fmaxf(mx, v);
      }
#pragma unroll
      for (int msk = 1; msk < 16; msk <<= 1)
        mx = fmaxf(mx, __shfl_xor(mx, msk, 32));
      bmax[r] = mx;
    }
#pragma unroll
    for (int r = 0; r < 8; ++r) {
      float mnew = fmaxf(m[r], bmax[r]);
      float corr = __expf(m[r] - mnew);
      float rsum = 0.f;
#pragma unroll
      for (int ns = 0; ns < 4; ++ns) {
        float p = __expf(s[ns][r] - mnew);
        s[ns][r] = p;
        rsum += p;
      }
#pragma unroll
      for (int msk = 1; msk < 16; msk <<= 1)
        rsum += __shfl_xor(rsum, msk, 32);
      l[r] = l[r] * corr + rsum;
      m[r] = mnew;
#pragma unroll
      for (int ns = 0; ns < 4; ++ns) o[ns][r] *= corr;
    }

    // Stage P (per-wave; DS ops within a wave are in order -> no barrier).
#pragma unroll
    for (int ns = 0; ns < 4; ++ns)
#pragma unroll
      for (int r = 0; r < 8; ++r)
        Ps[wave][r + 8 * hh][ns * 16 + col] = (_Float16)s[ns][r];

    // O += P * V  (A = P[16 x 64 keys], B = V[64 keys x 64 hd] via Vs^T)
#pragma unroll
    for (int ns = 0; ns < 4; ++ns)
#pragma unroll
      for (int kk = 0; kk < 2; ++kk) {
        v16h a = frag_a16(&Ps[wave][0][kk * 32], 72, lane);
        v16h bb = frag_b16(&Vs[ns * 16][kk * 32], 72, lane);
        o[ns] = wmma_f16(a, bb, o[ns]);
      }
  }

#pragma unroll
  for (int ns = 0; ns < 4; ++ns)
#pragma unroll
    for (int r = 0; r < 8; ++r) {
      int qrow = qbase + wave * 16 + r + 8 * hh;
      float v = o[ns][r] / l[r];
      Oh[(rowbase + qrow) * Dd + h * HD + ns * 16 + col] = (_Float16)v;
    }
}

// ---------------------------------------------------------------------------
// Host-side launch sequence (graph-capture safe: only kernel launches).
// ---------------------------------------------------------------------------
extern "C" void kernel_launch(void* const* d_in, const int* in_sizes, int n_in,
                              void* d_out, int out_size, void* d_ws,
                              size_t ws_size, hipStream_t stream) {
  (void)in_sizes; (void)n_in; (void)out_size; (void)ws_size;
  const int V = 32000, D = 1024, H = 16, HD = 64, L = 8, B = 4, T = 1024;
  const int Mrows = B * T;  // 4096

  const int*   idx     = (const int*)d_in[0];
  const float* tok_emb = (const float*)d_in[1];
  const float* pos_emb = (const float*)d_in[2];
  const float* Wq      = (const float*)d_in[3];   // [L,H,D,HD]
  const float* Wk      = (const float*)d_in[4];
  const float* Wv      = (const float*)d_in[5];
  const float* Wo      = (const float*)d_in[6];   // [L,D,D]
  const float* bo      = (const float*)d_in[7];   // [L,D]
  const float* ln1_g   = (const float*)d_in[8];
  const float* ln1_b   = (const float*)d_in[9];
  const float* ln2_g   = (const float*)d_in[10];
  const float* ln2_b   = (const float*)d_in[11];
  const float* W1      = (const float*)d_in[12];  // [L,D,4D]
  const float* b1      = (const float*)d_in[13];  // [L,4D]
  const float* W2      = (const float*)d_in[14];  // [L,4D,D]
  const float* b2      = (const float*)d_in[15];  // [L,D]
  const float* lnf_g   = (const float*)d_in[16];
  const float* lnf_b   = (const float*)d_in[17];
  const float* Whead   = (const float*)d_in[18];  // [D,V]
  const float* bhead   = (const float*)d_in[19];  // [V]

  // Workspace layout (~88 MB total).
  char* ws = (char*)d_ws;
  float*     x  = (float*)ws;                              // 16 MB fp32 [4096,1024]
  _Float16*  xn = (_Float16*)(ws + (16u << 20));           //  8 MB f16
  _Float16*  qh = (_Float16*)(ws + (24u << 20));           //  8 MB f16
  _Float16*  kh = (_Float16*)(ws + (32u << 20));           //  8 MB f16
  _Float16*  vh = (_Float16*)(ws + (40u << 20));           //  8 MB f16
  _Float16*  ah = (_Float16*)(ws + (48u << 20));           //  8 MB f16
  _Float16*  hh = (_Float16*)(ws + (56u << 20));           // 32 MB f16 [4096,4096]

  dim3 blk(256);
  embed_kernel<<<Mrows, blk, 0, stream>>>(idx, tok_emb, pos_emb, x, T, D);

  dim3 gProj(Mrows / 128, D / 128);          // 32 x 8
  dim3 gMlp1(Mrows / 128, (4 * D) / 128);    // 32 x 32
  dim3 gAttn(T / 128, H, B);                 // 8 x 16 x 4
  dim3 gHead(Mrows / 128, V / 128);          // 32 x 250

  for (int layer = 0; layer < L; ++layer) {
    const float* wq = Wq + (size_t)layer * H * D * HD;
    const float* wk = Wk + (size_t)layer * H * D * HD;
    const float* wv = Wv + (size_t)layer * H * D * HD;
    const float* wo = Wo + (size_t)layer * D * D;
    const float* w1 = W1 + (size_t)layer * D * 4 * D;
    const float* w2 = W2 + (size_t)layer * 4 * D * D;

    layernorm_kernel<<<Mrows, blk, 0, stream>>>(
        x, ln1_g + (size_t)layer * D, ln1_b + (size_t)layer * D, xn, D);

    // QKV projections (headed weight layout), f16 outputs.
    gemm_wmma<true, false, false, false, true><<<gProj, blk, 0, stream>>>(
        xn, wq, nullptr, nullptr, nullptr, qh, Mrows, D, D);
    gemm_wmma<true, false, false, false, true><<<gProj, blk, 0, stream>>>(
        xn, wk, nullptr, nullptr, nullptr, kh, Mrows, D, D);
    gemm_wmma<true, false, false, false, true><<<gProj, blk, 0, stream>>>(
        xn, wv, nullptr, nullptr, nullptr, vh, Mrows, D, D);

    attention_kernel<<<gAttn, blk, 0, stream>>>(qh, kh, vh, ah, T, D);

    // x = x + attn_out @ Wo + bo   (fp32, in-place residual)
    gemm_wmma<false, true, true, false, false><<<gProj, blk, 0, stream>>>(
        ah, wo, bo + (size_t)layer * D, x, x, nullptr, Mrows, D, D);

    layernorm_kernel<<<Mrows, blk, 0, stream>>>(
        x, ln2_g + (size_t)layer * D, ln2_b + (size_t)layer * D, xn, D);

    // h = gelu(xn @ W1 + b1), f16
    gemm_wmma<false, true, false, true, true><<<gMlp1, blk, 0, stream>>>(
        xn, w1, b1 + (size_t)layer * 4 * D, nullptr, nullptr, hh,
        Mrows, 4 * D, D);

    // x = x + h @ W2 + b2   (fp32, in-place residual)
    gemm_wmma<false, true, true, false, false><<<gProj, blk, 0, stream>>>(
        hh, w2, b2 + (size_t)layer * D, x, x, nullptr, Mrows, D, 4 * D);
  }

  layernorm_kernel<<<Mrows, blk, 0, stream>>>(x, lnf_g, lnf_b, xn, D);

  // logits = xn @ Whead + bhead  -> d_out (fp32)
  gemm_wmma<false, true, false, false, false><<<gHead, blk, 0, stream>>>(
      xn, Whead, bhead, nullptr, (float*)d_out, nullptr, Mrows, V, D);
}
